// DiscreteTimeNeuralGraph_65747359367615
// MI455X (gfx1250) — matrix operand order repaired and checked by reading, not compiled, not run on hardware
//
#include <hip/hip_runtime.h>
#include <hip/hip_bf16.h>

typedef __attribute__((ext_vector_type(16))) _Float16 v16h;
typedef __attribute__((ext_vector_type(8)))  float    v8f;
typedef __attribute__((ext_vector_type(2)))  _Float16 h2v;

#define DIMC   512
#define DSC    128
#define FEATC  256
#define NLAYER 8
#define BATCH  32
#define HW32   1024   // 32*32
#define OUTC   1000
#define KEEP_K 26214  // int(0.1 * 512 * 512)

// ---------------------------------------------------------------- stem conv1
// x:(32,3,128,128) -> out1:(32,64,64,64) [b][c][hw4096], stride2 pad1, +bias
__global__ __launch_bounds__(256) void k_conv1(const float* __restrict__ x,
                                               const float* __restrict__ w1,
                                               const float* __restrict__ b1,
                                               float* __restrict__ out1) {
  unsigned t = blockIdx.x * 256u + threadIdx.x;        // 32*64*4096
  int ox = t & 63, oy = (t >> 6) & 63, c = (t >> 12) & 63, b = t >> 18;
  float acc = b1[c];
  #pragma unroll
  for (int ic = 0; ic < 3; ++ic)
    #pragma unroll
    for (int ky = 0; ky < 3; ++ky) {
      int iy = 2 * oy - 1 + ky;
      if (iy < 0 || iy >= 128) continue;
      #pragma unroll
      for (int kx = 0; kx < 3; ++kx) {
        int ix = 2 * ox - 1 + kx;
        if (ix < 0 || ix >= 128) continue;
        acc += x[(((size_t)b * 3 + ic) * 128 + iy) * 128 + ix] *
               w1[((c * 3 + ic) * 3 + ky) * 3 + kx];
      }
    }
  out1[(((size_t)b * 64 + c) << 12) + (oy << 6) + ox] = acc;
}

// ------------------------------------------------- per-channel batch stats
// X:[b][c][HW] fp32 -> mean[c], rstd[c]; one block per channel
__global__ __launch_bounds__(256) void k_bnstat(const float* __restrict__ X,
                                                int B, int HW, int C,
                                                float* __restrict__ mean,
                                                float* __restrict__ rstd) {
  __shared__ float s1[256], s2[256];
  int c = blockIdx.x, tid = threadIdx.x;
  int n = B * HW;
  float a = 0.f, q = 0.f;
  for (int i = tid; i < n; i += 256) {
    int b = i / HW, p = i - b * HW;
    float v = X[((size_t)b * C + c) * HW + p];
    a += v; q += v * v;
  }
  s1[tid] = a; s2[tid] = q; __syncthreads();
  for (int s = 128; s > 0; s >>= 1) {
    if (tid < s) { s1[tid] += s1[tid + s]; s2[tid] += s2[tid + s]; }
    __syncthreads();
  }
  if (tid == 0) {
    float m = s1[0] / n, var = s2[0] / n - m * m;
    mean[c] = m; rstd[c] = rsqrtf(var + 1e-5f);
  }
}

// ------------------------------- stem: BN1+relu fused into depthwise s2 conv
// out1:(32,64,64,64) -> h2:(32,64,32,32)
__global__ __launch_bounds__(256) void k_dw2(const float* __restrict__ out1,
                                             const float* __restrict__ m1,
                                             const float* __restrict__ r1,
                                             const float* __restrict__ g1,
                                             const float* __restrict__ be1,
                                             const float* __restrict__ w2,
                                             const float* __restrict__ b2,
                                             float* __restrict__ h2) {
  unsigned t = blockIdx.x * 256u + threadIdx.x;        // 32*64*1024
  int ox = t & 31, oy = (t >> 5) & 31, c = (t >> 10) & 63, b = t >> 16;
  float sc = g1[c] * r1[c];
  float sh = be1[c] - m1[c] * sc;
  float acc = b2[c];
  const float* P = out1 + (((size_t)b * 64 + c) << 12);
  #pragma unroll
  for (int ky = 0; ky < 3; ++ky) {
    int iy = 2 * oy - 1 + ky;
    if (iy < 0 || iy >= 64) continue;
    #pragma unroll
    for (int kx = 0; kx < 3; ++kx) {
      int ix = 2 * ox - 1 + kx;
      if (ix < 0 || ix >= 64) continue;
      float v = P[(iy << 6) + ix] * sc + sh;
      v = v > 0.f ? v : 0.f;
      acc += v * w2[c * 9 + ky * 3 + kx];
    }
  }
  h2[(((size_t)b * 64 + c) << 10) + (oy << 5) + ox] = acc;
}

// ---------------------------------------- stem pointwise 64 -> 128 (+bias)
__global__ __launch_bounds__(256) void k_pw3(const float* __restrict__ h2,
                                             const float* __restrict__ w3,
                                             const float* __restrict__ b3,
                                             float* __restrict__ h3) {
  unsigned t = blockIdx.x * 256u + threadIdx.x;        // 32*128*1024
  int hw = t & 1023, o = (t >> 10) & 127, b = t >> 17;
  float acc = b3[o];
  const float* base = h2 + (((size_t)b * 64) << 10) + hw;
  #pragma unroll 8
  for (int i = 0; i < 64; ++i)
    acc += w3[o * 64 + i] * base[(size_t)i << 10];
  h3[(((size_t)b * 128 + o) << 10) + hw] = acc;
}

// ---------------- build cur0: BN2 on first 128 channels, zeros for 128..511
__global__ __launch_bounds__(256) void k_cur0(const float* __restrict__ h3,
                                              const float* __restrict__ m2,
                                              const float* __restrict__ r2,
                                              const float* __restrict__ g2,
                                              const float* __restrict__ be2,
                                              float* __restrict__ cur) {
  unsigned t = blockIdx.x * 256u + threadIdx.x;        // 32*512*1024
  int hw = t & 1023, c = (t >> 10) & 511, b = t >> 19;
  float v = 0.f;
  if (c < 128) {
    float x = h3[(((size_t)b * 128 + c) << 10) + hw];
    v = g2[c] * (x - m2[c]) * r2[c] + be2[c];
  }
  cur[t] = v;
}

// ------------------------ top-k |w| threshold via bisection (single block)
__global__ __launch_bounds__(1024) void k_thresh(const float* __restrict__ w,
                                                 float* __restrict__ out) {
  __shared__ float sf[1024];
  __shared__ int   si[1024];
  __shared__ float bounds[2];
  int tid = threadIdx.x;
  float mx = 0.f;
  for (int i = tid; i < 262144; i += 1024) mx = fmaxf(mx, fabsf(w[i]));
  sf[tid] = mx; __syncthreads();
  for (int s = 512; s > 0; s >>= 1) {
    if (tid < s) sf[tid] = fmaxf(sf[tid], sf[tid + s]);
    __syncthreads();
  }
  if (tid == 0) { bounds[0] = 0.f; bounds[1] = sf[0]; }
  __syncthreads();
  for (int it = 0; it < 28; ++it) {
    float mid = 0.5f * (bounds[0] + bounds[1]);
    int cnt = 0;
    for (int i = tid; i < 262144; i += 1024) cnt += (fabsf(w[i]) >= mid);
    si[tid] = cnt; __syncthreads();
    for (int s = 512; s > 0; s >>= 1) {
      if (tid < s) si[tid] += si[tid + s];
      __syncthreads();
    }
    if (tid == 0) { if (si[0] >= KEEP_K) bounds[0] = mid; else bounds[1] = mid; }
    __syncthreads();
  }
  if (tid == 0) out[0] = bounds[0];
}

// ------------------------------------------ materialize pruned w_eff in f16
__global__ __launch_bounds__(256) void k_weff(const float* __restrict__ w,
                                              const float* __restrict__ thr,
                                              _Float16* __restrict__ we) {
  unsigned t = blockIdx.x * 256u + threadIdx.x;        // 262144
  float v = w[t];
  we[t] = (fabsf(v) >= thr[0]) ? (_Float16)v : (_Float16)0.f;
}

// ------------------- layer: relu + depthwise 3x3 (NCHW f32 -> NHWC f16)
__global__ __launch_bounds__(256) void k_dwrelu(const float* __restrict__ cur,
                                                const float* __restrict__ cw,
                                                _Float16* __restrict__ T) {
  unsigned t = blockIdx.x * 256u + threadIdx.x;        // 32*512*1024
  int x = t & 31, y = (t >> 5) & 31, c = (t >> 10) & 511, b = t >> 19;
  const float* P = cur + (((size_t)b * DIMC + c) << 10);
  float acc = 0.f;
  #pragma unroll
  for (int ky = 0; ky < 3; ++ky) {
    int yy = y + ky - 1;
    if (yy < 0 || yy >= 32) continue;
    #pragma unroll
    for (int kx = 0; kx < 3; ++kx) {
      int xx = x + kx - 1;
      if (xx < 0 || xx >= 32) continue;
      float v = P[(yy << 5) + xx];
      v = v > 0.f ? v : 0.f;
      acc += v * cw[c * 9 + ky * 3 + kx];
    }
  }
  T[((size_t)((b << 10) + (y << 5) + x)) * DIMC + c] = (_Float16)acc;
}

// --------------------------- channel-mix GEMM via v_wmma_f32_16x16x32_f16
// W: (512 x 512) f16 row-major; X: (32768 x 512) f16 (NHWC); Y: [b][o][hw] f32
// Each wave: one 32(M) x 64(N) strip (2 M-tiles x 4 N-tiles); every B fragment
// feeds two WMMAs. Grid sized so every wave is full (EXEC all-1s for WMMA).
__global__ __launch_bounds__(128) void k_gemm(const _Float16* __restrict__ W,
                                              const _Float16* __restrict__ X,
                                              float* __restrict__ Y) {
  const int gw   = (int)(blockIdx.x * 4u + (threadIdx.x >> 5));
  const int lane = threadIdx.x & 31;
  const int b    = gw >> 8;            // 256 waves per batch (16 Mt x 16 Nt)
  const int rem  = gw & 255;
  const int m0   = (rem >> 4) << 5;    // 0..480, 32-row strip
  const int nb   = (rem & 15) << 6;    // 0..960
  const int hf   = lane >> 4;
  const int l16  = lane & 15;

  v8f acc[2][4] = {};
  const _Float16* wrow0 = W + (size_t)(m0 + l16) * DIMC;
  const _Float16* wrow1 = wrow0 + ((size_t)16 * DIMC);

  for (int k0 = 0; k0 < DIMC; k0 += 32) {
    v16h a0, a1;
    #pragma unroll
    for (int v = 0; v < 8; ++v) {
      int kk = k0 + ((v < 4) ? 0 : 16) + 8 * hf + 2 * (v & 3);  // ISA A layout
      h2v p0 = *(const h2v*)(wrow0 + kk);
      h2v p1 = *(const h2v*)(wrow1 + kk);
      a0[2 * v] = p0.x; a0[2 * v + 1] = p0.y;
      a1[2 * v] = p1.x; a1[2 * v + 1] = p1.y;
    }
    #pragma unroll
    for (int j = 0; j < 4; ++j) {
      const int n = nb + j * 16 + l16;
      const _Float16* xrow = X + ((size_t)((b << 10) + n)) * DIMC;
      v16h bf;
      #pragma unroll
      for (int v = 0; v < 8; ++v) {
        int kk = k0 + ((v < 4) ? 0 : 16) + 8 * hf + 2 * (v & 3);  // ISA B layout
        h2v p = *(const h2v*)(xrow + kk);
        bf[2 * v] = p.x; bf[2 * v + 1] = p.y;
      }
      acc[0][j] = __builtin_amdgcn_wmma_f32_16x16x32_f16(
          false, a0, false, bf, (short)0, acc[0][j], false, false);
      acc[1][j] = __builtin_amdgcn_wmma_f32_16x16x32_f16(
          false, a1, false, bf, (short)0, acc[1][j], false, false);
    }
  }
  // store D (f32 16x16 layout: VGPR v -> row base + v + 8*hf, col = n)
  #pragma unroll
  for (int mt = 0; mt < 2; ++mt) {
    #pragma unroll
    for (int j = 0; j < 4; ++j) {
      const int n = nb + j * 16 + l16;
      float* yb = Y + ((size_t)b * DIMC << 10) + n;
      #pragma unroll
      for (int v = 0; v < 8; ++v)
        yb[(size_t)(m0 + mt * 16 + v + 8 * hf) << 10] = acc[mt][j][v];
    }
  }
}

// ----------------------------------- instance norm: one block per (b, c)
__global__ __launch_bounds__(256) void k_inorm(const float* __restrict__ X,
                                               const float* __restrict__ G,
                                               const float* __restrict__ Bt,
                                               float* __restrict__ Y) {
  __shared__ float s1[256], s2[256], mb[2];
  int bc = blockIdx.x, tid = threadIdx.x;
  int c = bc & 511;
  const float* xp = X + ((size_t)bc << 10);
  float vals[4], a = 0.f, q = 0.f;
  #pragma unroll
  for (int j = 0; j < 4; ++j) {
    float v = xp[tid + (j << 8)];
    vals[j] = v; a += v; q += v * v;
  }
  s1[tid] = a; s2[tid] = q; __syncthreads();
  for (int s = 128; s > 0; s >>= 1) {
    if (tid < s) { s1[tid] += s1[tid + s]; s2[tid] += s2[tid + s]; }
    __syncthreads();
  }
  if (tid == 0) {
    float m = s1[0] * (1.f / 1024.f);
    float var = s2[0] * (1.f / 1024.f) - m * m;
    mb[0] = m; mb[1] = rsqrtf(var + 1e-5f);
  }
  __syncthreads();
  float gg = G[c] * mb[1];
  float bb = Bt[c] - mb[0] * gg;
  float* yp = Y + ((size_t)bc << 10);
  #pragma unroll
  for (int j = 0; j < 4; ++j) yp[tid + (j << 8)] = vals[j] * gg + bb;
}

// --------------------------- 2x2 mean at rows/cols {14,15}, last 256 chans
__global__ __launch_bounds__(256) void k_pool(const float* __restrict__ cur,
                                              float* __restrict__ pooled) {
  unsigned t = blockIdx.x * 256u + threadIdx.x;        // 32*256
  int f = t & 255, b = t >> 8;
  const float* base = cur + (((size_t)b * DIMC + 256 + f) << 10);
  pooled[t] = 0.25f * (base[14 * 32 + 14] + base[14 * 32 + 15] +
                       base[15 * 32 + 14] + base[15 * 32 + 15]);
}

__global__ __launch_bounds__(256) void k_fc(const float* __restrict__ pooled,
                                            const float* __restrict__ fw,
                                            const float* __restrict__ fb,
                                            float* __restrict__ out) {
  unsigned t = blockIdx.x * 256u + threadIdx.x;
  if (t >= BATCH * OUTC) return;
  int j = t % OUTC, b = t / OUTC;
  float acc = fb[j];
  const float* p = pooled + b * 256;
  #pragma unroll 8
  for (int f = 0; f < 256; ++f) acc += p[f] * fw[j * 256 + f];
  out[(size_t)b * OUTC + j] = acc;
}

extern "C" void kernel_launch(void* const* d_in, const int* in_sizes, int n_in,
                              void* d_out, int out_size, void* d_ws, size_t ws_size,
                              hipStream_t stream) {
  (void)in_sizes; (void)n_in; (void)out_size; (void)ws_size;
  const float* x      = (const float*)d_in[0];
  const float* ds_w1  = (const float*)d_in[1];
  const float* ds_b1  = (const float*)d_in[2];
  const float* bn1_g  = (const float*)d_in[3];
  const float* bn1_b  = (const float*)d_in[4];
  const float* ds_w2  = (const float*)d_in[5];
  const float* ds_b2  = (const float*)d_in[6];
  const float* ds_w3  = (const float*)d_in[7];
  const float* ds_b3  = (const float*)d_in[8];
  const float* bn2_g  = (const float*)d_in[9];
  const float* bn2_b  = (const float*)d_in[10];
  const float* conv_w = (const float*)d_in[11];
  const float* graphw = (const float*)d_in[12];
  const float* norm_g = (const float*)d_in[13];
  const float* norm_b = (const float*)d_in[14];
  const float* fc_w   = (const float*)d_in[15];
  const float* fc_b   = (const float*)d_in[16];
  float* out = (float*)d_out;

  const size_t MB = 1ull << 20;
  char* ws = (char*)d_ws;
  float*    cur    = (float*)ws;                          // 64 MB  [b][c][hw] f32
  float*    gbuf   = (float*)(ws + 64 * MB);              // 64 MB  gemm out / stem conv1
  _Float16* tmp    = (_Float16*)(ws + 128 * MB);          // 32 MB  NHWC f16 activations
  float*    h2     = (float*)(ws + 128 * MB);             // alias: stem (8 MB)
  float*    h3     = (float*)(ws + 136 * MB);             // alias: stem (16 MB)
  _Float16* weff   = (_Float16*)(ws + 160 * MB);          // 512 KB pruned graph_w f16
  float*    pooled = (float*)(ws + 160 * MB + 512 * 1024);
  float*    stats  = pooled + BATCH * FEATC;              // m1[64] r1[64] m2[128] r2[128] thr[1]
  float* m1 = stats, *r1 = stats + 64, *m2 = stats + 128, *r2 = stats + 256, *thr = stats + 384;

  // ---- stem ----
  k_conv1<<<(32 * 64 * 4096) / 256, 256, 0, stream>>>(x, ds_w1, ds_b1, gbuf);
  k_bnstat<<<64, 256, 0, stream>>>(gbuf, 32, 4096, 64, m1, r1);
  k_dw2<<<(32 * 64 * 1024) / 256, 256, 0, stream>>>(gbuf, m1, r1, bn1_g, bn1_b, ds_w2, ds_b2, h2);
  k_pw3<<<(32 * 128 * 1024) / 256, 256, 0, stream>>>(h2, ds_w3, ds_b3, h3);
  k_bnstat<<<128, 256, 0, stream>>>(h3, 32, 1024, 128, m2, r2);
  k_cur0<<<(32 * 512 * 1024) / 256, 256, 0, stream>>>(h3, m2, r2, bn2_g, bn2_b, cur);

  // ---- prune graph weights ----
  k_thresh<<<1, 1024, 0, stream>>>(graphw, thr);
  k_weff<<<262144 / 256, 256, 0, stream>>>(graphw, thr, weff);

  // ---- 8 graph layers: relu+dwconv -> WMMA channel mix -> instance norm ----
  for (int l = 0; l < NLAYER; ++l) {
    k_dwrelu<<<(32 * 512 * 1024) / 256, 256, 0, stream>>>(cur, conv_w, tmp);
    k_gemm<<<2048, 128, 0, stream>>>(weff, tmp, gbuf);   // 8192 full waves
    k_inorm<<<32 * 512, 256, 0, stream>>>(gbuf, norm_g, norm_b, cur);
  }

  // ---- head ----
  k_pool<<<(32 * 256) / 256, 256, 0, stream>>>(cur, pooled);
  k_fc<<<(BATCH * OUTC + 255) / 256, 256, 0, stream>>>(pooled, fc_w, fc_b, out);
}
